// LRU_64312840290631
// MI455X (gfx1250) — compile-verified, matrix operand order checked by reading, not run on hardware
//
#include <hip/hip_runtime.h>
#include <hip/hip_bf16.h>
#include <math.h>

// ---------------------------------------------------------------------------
// Problem constants (from reference): L=4, T=32768, H=256, R=256, IN=OUT=256
// ---------------------------------------------------------------------------
#define TT 32768
#define HH 256
#define RR 256
#define LL 4

typedef __bf16 bf16_t;
typedef bf16_t v16bf __attribute__((ext_vector_type(16)));
typedef bf16_t v8bf  __attribute__((ext_vector_type(8)));
typedef float  v8f   __attribute__((ext_vector_type(8)));

// Native converts: clang lowers fptrunc float->bf16 to v_cvt_pk_bf16_f32 on
// gfx1250 (round-to-nearest-even), replacing the 4-op integer emulation.
__device__ __forceinline__ bf16_t f2bf(float f) { return (bf16_t)f; }

// ---------------------------------------------------------------------------
// WMMA GEMM:  out[M,N] = A[M,K] @ W[N,K]^T  (+bias, +scale[n]*extra[m,n], leaky)
// A is f32 (optionally stored transposed: A[k*lda+m]); W is pre-converted bf16
// row-major [N][K].  Block tile 128x128, 8 waves, each wave 32x64 (2x4 WMMA
// tiles of 16x16), K-step 32, bf16 tiles staged in LDS.
// ---------------------------------------------------------------------------
template <int TRANS_A, int TRANS_OUT, int HAS_BIAS, int LEAKY, int ADDSC>
__global__ __launch_bounds__(256) void gemm_wmma_kernel(
    const float* __restrict__ A, int lda,
    const bf16_t* __restrict__ W,           // [N][K] bf16
    const float* __restrict__ bias,         // [N] or null
    const float* __restrict__ scale,        // [N] or null   (D)
    const float* __restrict__ extra, int lde,  // [M][lde] or null (layer input)
    float* __restrict__ out, int ldo,
    int K) {
  __shared__ bf16_t As[128][40];   // 80B row stride: 16B-aligned frag loads
  __shared__ bf16_t Ws[128][40];

  const int tid  = threadIdx.x;
  const int lane = tid & 31;
  const int wid  = tid >> 5;
  const int m0   = blockIdx.x * 128;
  const int n0   = blockIdx.y * 128;
  const int wm   = (wid & 3) * 32;   // wave M offset within block tile
  const int wn   = (wid >> 2) * 64;  // wave N offset within block tile

  v8f acc[2][4];
#pragma unroll
  for (int mi = 0; mi < 2; ++mi)
#pragma unroll
    for (int ni = 0; ni < 4; ++ni)
#pragma unroll
      for (int i = 0; i < 8; ++i) acc[mi][ni][i] = 0.0f;

  for (int k0 = 0; k0 < K; k0 += 32) {
    // ---- issue all staging loads first (4x A float4, 2x W uint4) ----
    float4 fA[4];
    uint4  wV[2];
#pragma unroll
    for (int j = 0; j < 4; ++j) {
      const int v = tid + 256 * j;       // 1024 float4 chunks of the A tile
      if constexpr (TRANS_A) {
        const int kk = v >> 5;           // k row within tile (0..31)
        const int mq = v & 31;           // 4-wide m chunk (0..31)
        fA[j] = *(const float4*)(A + (size_t)(k0 + kk) * lda + m0 + mq * 4);
      } else {
        const int row = v >> 3;          // m row (0..127), 8 float4 per row
        const int kq  = v & 7;
        fA[j] = *(const float4*)(A + (size_t)(m0 + row) * lda + k0 + kq * 4);
      }
    }
#pragma unroll
    for (int j = 0; j < 2; ++j) {
      const int c   = tid + 256 * j;     // 512 chunks of 8 bf16 (16B)
      const int row = c >> 2;
      const int kq  = c & 3;
      wV[j] = *(const uint4*)(W + (size_t)(n0 + row) * K + k0 + kq * 8);
    }
    if (k0 + 32 < K) {
      const float* pf = TRANS_A ? (A + (size_t)(k0 + 32) * lda + m0)
                                : (A + (size_t)m0 * lda + k0 + 32);
      __builtin_prefetch(pf, 0, 3);      // global_prefetch_b8
    }

    // ---- convert + stage into LDS ----
#pragma unroll
    for (int j = 0; j < 4; ++j) {
      const int v = tid + 256 * j;
      if constexpr (TRANS_A) {
        const int kk = v >> 5;
        const int mq = v & 31;
        As[mq * 4 + 0][kk] = f2bf(fA[j].x);
        As[mq * 4 + 1][kk] = f2bf(fA[j].y);
        As[mq * 4 + 2][kk] = f2bf(fA[j].z);
        As[mq * 4 + 3][kk] = f2bf(fA[j].w);
      } else {
        const int row = v >> 3;
        const int kq  = v & 7;
        As[row][kq * 4 + 0] = f2bf(fA[j].x);
        As[row][kq * 4 + 1] = f2bf(fA[j].y);
        As[row][kq * 4 + 2] = f2bf(fA[j].z);
        As[row][kq * 4 + 3] = f2bf(fA[j].w);
      }
    }
#pragma unroll
    for (int j = 0; j < 2; ++j) {
      const int c   = tid + 256 * j;
      const int row = c >> 2;
      const int kq  = c & 3;
      *(uint4*)&Ws[row][kq * 8] = wV[j];
    }
    __syncthreads();

    // ---- build fragments and issue WMMAs ----
    const int kof = (lane >> 4) * 16;    // lanes 0-15: K 0..15, lanes 16-31: K 16..31
    v16bf af[2], bfg[4];
#pragma unroll
    for (int mi = 0; mi < 2; ++mi) {
      const int r = wm + mi * 16 + (lane & 15);
      const v8bf* p = (const v8bf*)&As[r][kof];
      v8bf lo = p[0], hi = p[1];
#pragma unroll
      for (int i = 0; i < 8; ++i) { af[mi][i] = lo[i]; af[mi][8 + i] = hi[i]; }
    }
#pragma unroll
    for (int ni = 0; ni < 4; ++ni) {
      const int r = wn + ni * 16 + (lane & 15);
      const v8bf* p = (const v8bf*)&Ws[r][kof];
      v8bf lo = p[0], hi = p[1];
#pragma unroll
      for (int i = 0; i < 8; ++i) { bfg[ni][i] = lo[i]; bfg[ni][8 + i] = hi[i]; }
    }
#pragma unroll
    for (int mi = 0; mi < 2; ++mi)
#pragma unroll
      for (int ni = 0; ni < 4; ++ni)
        acc[mi][ni] = __builtin_amdgcn_wmma_f32_16x16x32_bf16(
            false, af[mi], false, bfg[ni], (short)0, acc[mi][ni], false, false);
    __syncthreads();
  }

  // ---- epilogue ----
#pragma unroll
  for (int mi = 0; mi < 2; ++mi)
#pragma unroll
    for (int ni = 0; ni < 4; ++ni) {
      const int nG    = n0 + wn + ni * 16 + (lane & 15);
      const int mBase = m0 + wm + mi * 16 + ((lane >> 4) * 8);
#pragma unroll
      for (int i = 0; i < 8; ++i) {
        const int mG = mBase + i;
        float v = acc[mi][ni][i];
        if constexpr (HAS_BIAS) v += bias[nG];
        if constexpr (ADDSC)    v += scale[nG] * extra[(size_t)mG * lde + nG];
        if constexpr (LEAKY)    v = v > 0.0f ? v : 0.01f * v;
        if constexpr (TRANS_OUT) out[(size_t)nG * ldo + mG] = v;
        else                     out[(size_t)mG * ldo + nG] = v;
      }
    }
}

// ---------------------------------------------------------------------------
// Blocked associative scan:  s_t = A_t * s_{t-1} + b_t  (complex, A=0 at reset)
// One block per recurrent channel r; 256 threads scan chunks of 256 timesteps
// (shfl wave scan + LDS cross-wave combine + serial chunk carry).
// Bu_T / states_T are [2R][T]: row r = real, row R+r = imag.
// ---------------------------------------------------------------------------
__global__ __launch_bounds__(256) void lru_scan_kernel(
    const float* __restrict__ BuT,
    const unsigned char* __restrict__ start,
    const float* __restrict__ lamTab,   // [R][2] for this layer
    const float* __restrict__ h0,       // [R] for this layer
    float* __restrict__ statesT) {
  const int r    = blockIdx.x;
  const int tid  = threadIdx.x;
  const int lane = tid & 31;
  const int wid  = tid >> 5;
  const float lamr = lamTab[2 * r + 0];
  const float lami = lamTab[2 * r + 1];

  __shared__ float sAr[8], sAi[8], sbr[8], sbi[8];
  __shared__ float carry[4];

  float cAr = 1.0f, cAi = 0.0f, cbr = h0[r], cbi = 0.0f;

  const float* bre = BuT + (size_t)r * TT;
  const float* bim = BuT + (size_t)(RR + r) * TT;
  float* sre = statesT + (size_t)r * TT;
  float* sim = statesT + (size_t)(RR + r) * TT;

  for (int t0 = 0; t0 < TT; t0 += 256) {
    const int t = t0 + tid;
    float br_ = bre[t];
    float bi_ = bim[t];
    const int sf = start[t];
    float Ar = sf ? 0.0f : lamr;
    float Ai = sf ? 0.0f : lami;

    // intra-wave inclusive scan (Hillis-Steele) of (A, b) under complex combine
#pragma unroll
    for (int off = 1; off < 32; off <<= 1) {
      float pAr = __shfl_up(Ar, off);
      float pAi = __shfl_up(Ai, off);
      float pbr = __shfl_up(br_, off);
      float pbi = __shfl_up(bi_, off);
      if (lane >= off) {
        float nbr = Ar * pbr - Ai * pbi + br_;
        float nbi = Ar * pbi + Ai * pbr + bi_;
        float nAr = Ar * pAr - Ai * pAi;
        float nAi = Ar * pAi + Ai * pAr;
        Ar = nAr; Ai = nAi; br_ = nbr; bi_ = nbi;
      }
    }
    if (lane == 31) { sAr[wid] = Ar; sAi[wid] = Ai; sbr[wid] = br_; sbi[wid] = bi_; }
    __syncthreads();

    // prefix = carry ∘ aggregates of waves < wid
    float pAr = cAr, pAi = cAi, pbr = cbr, pbi = cbi;
    for (int w = 0; w < wid; ++w) {
      const float wAr = sAr[w], wAi = sAi[w], wbr = sbr[w], wbi = sbi[w];
      float nbr = wAr * pbr - wAi * pbi + wbr;
      float nbi = wAr * pbi + wAi * pbr + wbi;
      float nAr = wAr * pAr - wAi * pAi;
      float nAi = wAr * pAi + wAi * pAr;
      pAr = nAr; pAi = nAi; pbr = nbr; pbi = nbi;
    }
    const float fr = Ar * pbr - Ai * pbi + br_;
    const float fi = Ar * pbi + Ai * pbr + bi_;
    sre[t] = fr;
    sim[t] = fi;
    if (tid == 255) {
      carry[0] = Ar * pAr - Ai * pAi;
      carry[1] = Ar * pAi + Ai * pAr;
      carry[2] = fr;
      carry[3] = fi;
    }
    __syncthreads();
    cAr = carry[0]; cAi = carry[1]; cbr = carry[2]; cbi = carry[3];
  }
}

// ---------------------------------------------------------------------------
// Weight preparation kernels (f32 -> bf16, fold gamma into B, -C_im, lam table)
// ---------------------------------------------------------------------------
__global__ void conv_bf16_kernel(const float* __restrict__ src,
                                 bf16_t* __restrict__ dst, int n) {
  int i = blockIdx.x * 256 + threadIdx.x;
  if (i < n) dst[i] = f2bf(src[i]);
}

__global__ void prep_b_kernel(const float* __restrict__ Bre,
                              const float* __restrict__ Bim,
                              const float* __restrict__ gamma_log,
                              bf16_t* __restrict__ dst) {
  const int l = blockIdx.y;
  const int i = blockIdx.x * 256 + threadIdx.x;   // over R*H
  if (i >= RR * HH) return;
  const int r = i / HH, h = i % HH;
  const float g = __expf(gamma_log[l * RR + r]);
  bf16_t* d = dst + (size_t)l * (2 * RR * HH);    // [2R][H]
  d[(size_t)r * HH + h]        = f2bf(Bre[(size_t)l * RR * HH + i] * g);
  d[(size_t)(RR + r) * HH + h] = f2bf(Bim[(size_t)l * RR * HH + i] * g);
}

__global__ void prep_c_kernel(const float* __restrict__ Cre,
                              const float* __restrict__ Cim,
                              bf16_t* __restrict__ dst) {
  const int l = blockIdx.y;
  const int i = blockIdx.x * 256 + threadIdx.x;   // over H*R
  if (i >= HH * RR) return;
  const int h = i / RR, r = i % RR;
  bf16_t* d = dst + (size_t)l * (HH * 2 * RR);    // [H][2R]
  d[(size_t)h * (2 * RR) + r]      = f2bf(Cre[(size_t)l * HH * RR + i]);
  d[(size_t)h * (2 * RR) + RR + r] = f2bf(-Cim[(size_t)l * HH * RR + i]);
}

__global__ void prep_lam_kernel(const float* __restrict__ nu_log,
                                const float* __restrict__ theta_log,
                                float* __restrict__ lamTab) {
  int i = blockIdx.x * 256 + threadIdx.x;         // over L*R
  if (i >= LL * RR) return;
  const float mag = __expf(-__expf(nu_log[i]));
  const float ang = __expf(theta_log[i]);
  lamTab[2 * i + 0] = mag * __cosf(ang);
  lamTab[2 * i + 1] = mag * __sinf(ang);
}

// ---------------------------------------------------------------------------
// Host launcher
// ---------------------------------------------------------------------------
extern "C" void kernel_launch(void* const* d_in, const int* in_sizes, int n_in,
                              void* d_out, int out_size, void* d_ws,
                              size_t ws_size, hipStream_t stream) {
  (void)in_sizes; (void)n_in; (void)out_size; (void)ws_size;
  const float*         x         = (const float*)d_in[0];
  const unsigned char* start     = (const unsigned char*)d_in[1];
  const float*         h0        = (const float*)d_in[2];
  const float*         map_in_w  = (const float*)d_in[3];
  const float*         map_in_b  = (const float*)d_in[4];
  const float*         map_out_w = (const float*)d_in[5];
  const float*         map_out_b = (const float*)d_in[6];
  const float*         nu_log    = (const float*)d_in[7];
  const float*         theta_log = (const float*)d_in[8];
  const float*         gamma_log = (const float*)d_in[9];
  const float*         B_re      = (const float*)d_in[10];
  const float*         B_im      = (const float*)d_in[11];
  const float*         C_re      = (const float*)d_in[12];
  const float*         C_im      = (const float*)d_in[13];
  const float*         D         = (const float*)d_in[14];
  const float*         ffW       = (const float*)d_in[15];
  const float*         ffb       = (const float*)d_in[16];

  char* ws = (char*)d_ws;
  size_t off = 0;
  auto alloc = [&](size_t bytes) -> void* {
    void* p = ws + off;
    off += (bytes + 255) & ~(size_t)255;
    return p;
  };
  float* act0    = (float*)alloc((size_t)TT * HH * 4);
  float* act1    = (float*)alloc((size_t)TT * HH * 4);
  float* ybuf    = (float*)alloc((size_t)TT * HH * 4);
  float* BuT     = (float*)alloc((size_t)2 * RR * TT * 4);
  float* statesT = (float*)alloc((size_t)2 * RR * TT * 4);
  bf16_t* wInBf  = (bf16_t*)alloc((size_t)HH * HH * 2);
  bf16_t* wOutBf = (bf16_t*)alloc((size_t)HH * HH * 2);
  bf16_t* ffWBf  = (bf16_t*)alloc((size_t)LL * HH * HH * 2);
  bf16_t* bcatBf = (bf16_t*)alloc((size_t)LL * 2 * RR * HH * 2);
  bf16_t* ccatBf = (bf16_t*)alloc((size_t)LL * HH * 2 * RR * 2);
  float*  lamTab = (float*)alloc((size_t)LL * RR * 2 * 4);

  // --- weight prep ---
  conv_bf16_kernel<<<(HH * HH + 255) / 256, 256, 0, stream>>>(map_in_w, wInBf, HH * HH);
  conv_bf16_kernel<<<(HH * HH + 255) / 256, 256, 0, stream>>>(map_out_w, wOutBf, HH * HH);
  conv_bf16_kernel<<<(LL * HH * HH + 255) / 256, 256, 0, stream>>>(ffW, ffWBf, LL * HH * HH);
  prep_b_kernel<<<dim3((RR * HH + 255) / 256, LL), 256, 0, stream>>>(B_re, B_im, gamma_log, bcatBf);
  prep_c_kernel<<<dim3((HH * RR + 255) / 256, LL), 256, 0, stream>>>(C_re, C_im, ccatBf);
  prep_lam_kernel<<<(LL * RR + 255) / 256, 256, 0, stream>>>(nu_log, theta_log, lamTab);

  const dim3 gN1(TT / 128, HH / 128);       // N=256
  const dim3 gN2(TT / 128, (2 * RR) / 128); // N=512

  // --- input map: emb = x @ Win^T + b ---
  gemm_wmma_kernel<0, 0, 1, 0, 0><<<gN1, 256, 0, stream>>>(
      x, HH, wInBf, map_in_b, nullptr, nullptr, 0, act0, HH, HH);

  float* actIn = act0;
  float* actOut = act1;
  for (int l = 0; l < LL; ++l) {
    // Bu (re|im) = actIn @ [gamma*B_re ; gamma*B_im]^T, stored transposed [2R][T]
    gemm_wmma_kernel<0, 1, 0, 0, 0><<<gN2, 256, 0, stream>>>(
        actIn, HH, bcatBf + (size_t)l * 2 * RR * HH, nullptr, nullptr, nullptr, 0,
        BuT, TT, HH);
    // associative scan over T per channel
    lru_scan_kernel<<<RR, 256, 0, stream>>>(
        BuT, start, lamTab + (size_t)l * 2 * RR, h0 + (size_t)l * RR, statesT);
    // y = [s_re|s_im] @ [C_re | -C_im]^T + D*actIn  (A transposed, K=512)
    gemm_wmma_kernel<1, 0, 0, 0, 1><<<gN1, 256, 0, stream>>>(
        statesT, TT, ccatBf + (size_t)l * HH * 2 * RR, nullptr, D + (size_t)l * HH,
        actIn, HH, ybuf, HH, 2 * RR);
    // z = leaky_relu(y @ ffW^T + ffb)
    gemm_wmma_kernel<0, 0, 1, 1, 0><<<gN1, 256, 0, stream>>>(
        ybuf, HH, ffWBf + (size_t)l * HH * HH, ffb + (size_t)l * HH, nullptr,
        nullptr, 0, actOut, HH, HH);
    float* tmp = actIn; actIn = actOut; actOut = tmp;
  }

  // --- output map ---
  gemm_wmma_kernel<0, 0, 1, 0, 0><<<gN1, 256, 0, stream>>>(
      actIn, HH, wOutBf, map_out_b, nullptr, nullptr, 0, (float*)d_out, HH, HH);
}
